// MACLayer_833223655614
// MI455X (gfx1250) — matrix operand order repaired and checked by reading, not compiled, not run on hardware
//
#include <hip/hip_runtime.h>
#include <hip/hip_bf16.h>

// ---------------------------------------------------------------------------
// MAC / Titans layer forward for MI455X (gfx1250, wave32, WMMA).
// All GEMMs: v_wmma_f32_16x16x32_bf16 (bf16 inputs, f32 accumulate).
// B (and plain A) tiles staged into LDS with GLOBAL_LOAD_ASYNC_TO_LDS_B128
// (ASYNCcnt + s_wait_asynccnt), which matters most for the 604 MB fp32
// w_final stream (split-K + f32 global atomics, M=16 row tile).
// ---------------------------------------------------------------------------

typedef __bf16 bf16;
typedef __attribute__((ext_vector_type(16))) __bf16 v16bf;
typedef __attribute__((ext_vector_type(8)))  float   v8f;

union FragA { v16bf v; bf16 e[16]; };
union FragC { v8f  v; float f[8];  };

constexpr int B_  = 16,  S_ = 64, D_ = 128, P_ = 16;
constexpr int C_  = 144, FF_ = 512, I_ = 256, HD_ = 64;
constexpr int BS_ = 1024, BC_ = 2304, CD_ = 18432, SD_ = 8192;

__device__ __forceinline__ float siluf(float x) { return x / (1.0f + __expf(-x)); }
__device__ __forceinline__ float sigm(float x)  { return 1.0f / (1.0f + __expf(-x)); }

__device__ __forceinline__ unsigned lds_off32(const void* p)
{
  // LDS aperture: flat addr low 32 bits == LDS byte offset (ISA 10.2).
  return (unsigned)(size_t)p;
}

__device__ __forceinline__ void async_ld_b128(unsigned ldsoff, const float* gp)
{
  asm volatile("global_load_async_to_lds_b128 %0, %1, off"
               :: "v"(ldsoff), "v"((unsigned long long)(size_t)gp)
               : "memory");
}

__device__ __forceinline__ void wait_asynccnt0()
{
  asm volatile("s_wait_asynccnt 0x0" ::: "memory");
}

// --------------------------- generic WMMA GEMM -----------------------------
// C[M,N] = op(A) @ op(B) (+bias) (+silu) ; TA: A stored [K,M]; TB: B stored [N,K]
// ASILU: apply silu to A elements on load. EPI: 0 store, 1 +bias, 2 +bias+silu,
// 3 atomicAdd (split-K partials).
enum { EPI_NONE = 0, EPI_BIAS = 1, EPI_BIAS_SILU = 2, EPI_ATOMIC = 3 };

template<int TA, int TB, int ASILU, int EPI>
__global__ __launch_bounds__(256) void gemm_wmma(
    const float* __restrict__ A, const float* __restrict__ Bm,
    const float* __restrict__ bias, float* __restrict__ Cm,
    int M, int N, int K, int Kchunk)
{
  // Pads chosen for 16B-aligned B128 async chunks AND conflict-free frag reads:
  // A row stride 36 (144 B), B row stride 68 (272 B).
  __shared__ __align__(16) float As[32][36];
  __shared__ __align__(16) float Bs[32][68];
  const int tid  = threadIdx.x;
  const int lane = tid & 31;
  const int wave = tid >> 5;
  const int tm   = wave >> 2;        // 0..1
  const int tn   = wave & 3;         // 0..3
  const int bm   = blockIdx.y * 32;
  const int bn   = blockIdx.x * 64;
  const int k0   = blockIdx.z * Kchunk;
  const int k1   = (k0 + Kchunk < K) ? (k0 + Kchunk) : K;

  FragC acc;
#pragma unroll
  for (int i = 0; i < 8; ++i) acc.f[i] = 0.0f;

  for (int kk0 = k0; kk0 < k1; kk0 += 32) {
    // ---- stage A tile 32x32 (f32) ----
    if constexpr (!TA && !ASILU) {
      // async DMA path (all such launches have M%32==0, K%32==0: no guards)
      int e = tid * 4;                  // dword index in 32x32 tile
      int r = e >> 5, c = e & 31;       // c multiple of 4
      async_ld_b128(lds_off32(&As[r][c]), A + (size_t)(bm + r) * K + (kk0 + c));
    } else {
#pragma unroll
      for (int i = 0; i < 4; ++i) {
        int e = tid + i * 256;
        int r = e >> 5, c = e & 31;
        int gr = bm + r, gk = kk0 + c;
        float v = 0.0f;
        if (gr < M && gk < K)
          v = TA ? A[(size_t)gk * M + gr] : A[(size_t)gr * K + gk];
        if (ASILU) v = siluf(v);
        As[r][c] = v;
      }
    }
    // ---- stage B tile 32x64 (f32) ----
    if constexpr (!TB) {
      // async DMA path (all TB==0 launches have N%64==0, K%32==0: no guards)
#pragma unroll
      for (int i = 0; i < 2; ++i) {
        int e = (tid + i * 256) * 4;    // dword index in 32x64 tile
        int r = e >> 6, c = e & 63;     // c multiple of 4
        async_ld_b128(lds_off32(&Bs[r][c]), Bm + (size_t)(kk0 + r) * N + (bn + c));
      }
      wait_asynccnt0();
    } else {
#pragma unroll
      for (int i = 0; i < 8; ++i) {
        int e = tid + i * 256;
        int r = e >> 6, c = e & 63;
        int gk = kk0 + r, gn = bn + c;
        float v = 0.0f;
        if (gk < K && gn < N)
          v = Bm[(size_t)gn * K + gk];
        Bs[r][c] = v;
      }
      if constexpr (!TA && !ASILU) wait_asynccnt0();
    }
    __syncthreads();

    // build fragments per documented CDNA5 WMMA layouts
    FragA af, bf;
    const int mrow = tm * 16 + (lane & 15);
    const int kb   = (lane >> 4) * 8;
#pragma unroll
    for (int i = 0; i < 8; ++i) af.e[i]     = (bf16)As[mrow][kb + i];
#pragma unroll
    for (int i = 0; i < 8; ++i) af.e[8 + i] = (bf16)As[mrow][16 + kb + i];
    const int ncol = tn * 16 + (lane & 15);
    const int kb2  = (lane >> 4) * 16;
#pragma unroll
    for (int i = 0; i < 8; ++i) bf.e[i] = (bf16)Bs[kb + i][ncol];
#pragma unroll
    for (int i = 0; i < 8; ++i) bf.e[8 + i] = (bf16)Bs[16 + kb + i][ncol];

    acc.v = __builtin_amdgcn_wmma_f32_16x16x32_bf16(
        false, af.v, false, bf.v, (short)0, acc.v, false, false);
    __syncthreads();
  }

  const int half = lane >> 4;
  const int col  = bn + tn * 16 + (lane & 15);
  if (col < N) {
#pragma unroll
    for (int r = 0; r < 8; ++r) {
      int row = bm + tm * 16 + half * 8 + r;
      if (row < M) {
        float v = acc.f[r];
        if (EPI == EPI_BIAS || EPI == EPI_BIAS_SILU) v += bias[col];
        if (EPI == EPI_BIAS_SILU) v = siluf(v);
        if (EPI == EPI_ATOMIC) atomicAdd(&Cm[(size_t)row * N + col], v);
        else Cm[(size_t)row * N + col] = v;
      }
    }
  }
}

// --------------------------- attention kernels -----------------------------
// scores[b,h,t,s] = (Q_bh @ K_bh^T)/8 ; one wave per 16x16 tile, K=64
__global__ __launch_bounds__(32) void attn_scores(
    const float* __restrict__ qb, const float* __restrict__ kb_,
    float* __restrict__ sbuf)
{
  const int lane = threadIdx.x & 31;
  const int tn = blockIdx.x, tmI = blockIdx.y, bh = blockIdx.z;
  const int b = bh >> 1, h = bh & 1;
  FragC acc;
#pragma unroll
  for (int i = 0; i < 8; ++i) acc.f[i] = 0.0f;

  const int m = tmI * 16 + (lane & 15);
  const int n = tn * 16 + (lane & 15);
  const float* aptr = qb  + ((size_t)b * C_ + m) * D_ + h * HD_;
  const float* bptr = kb_ + ((size_t)b * C_ + n) * D_ + h * HD_;
  const int kbA = (lane >> 4) * 8;
  const int kbB = (lane >> 4) * 16;
#pragma unroll
  for (int kk0 = 0; kk0 < HD_; kk0 += 32) {
    FragA af, bf;
#pragma unroll
    for (int i = 0; i < 8; ++i) af.e[i]     = (bf16)aptr[kk0 + kbA + i];
#pragma unroll
    for (int i = 0; i < 8; ++i) af.e[8 + i] = (bf16)aptr[kk0 + 16 + kbA + i];
#pragma unroll
    for (int i = 0; i < 8; ++i) bf.e[i]     = (bf16)bptr[kk0 + kbA + i];
#pragma unroll
    for (int i = 0; i < 8; ++i) bf.e[8 + i] = (bf16)bptr[kk0 + 16 + kbA + i];
    acc.v = __builtin_amdgcn_wmma_f32_16x16x32_bf16(
        false, af.v, false, bf.v, (short)0, acc.v, false, false);
  }
  const int half = lane >> 4;
  const int col  = tn * 16 + (lane & 15);
  float* out = sbuf + (size_t)bh * C_ * C_;
#pragma unroll
  for (int r = 0; r < 8; ++r)
    out[(size_t)(tmI * 16 + half * 8 + r) * C_ + col] = acc.f[r] * 0.125f;
}

// O_bh[t,d] = A_bh @ V_bh ; K=144 padded to 160 with zero fill
__global__ __launch_bounds__(32) void attn_av(
    const float* __restrict__ sbuf, const float* __restrict__ vb,
    float* __restrict__ obuf)
{
  const int lane = threadIdx.x & 31;
  const int tn = blockIdx.x, tmI = blockIdx.y, bh = blockIdx.z;
  const int b = bh >> 1, h = bh & 1;
  FragC acc;
#pragma unroll
  for (int i = 0; i < 8; ++i) acc.f[i] = 0.0f;

  const int m = tmI * 16 + (lane & 15);
  const int n = tn * 16 + (lane & 15);
  const float* aptr = sbuf + ((size_t)bh * C_ + m) * C_;
  const int kbA = (lane >> 4) * 8;
  const int kbB = (lane >> 4) * 16;
#pragma unroll
  for (int kk0 = 0; kk0 < 160; kk0 += 32) {
    FragA af, bf;
#pragma unroll
    for (int i = 0; i < 8; ++i) {
      int k = kk0 + kbA + i;
      af.e[i] = (bf16)((k < C_) ? aptr[k] : 0.0f);
    }
#pragma unroll
    for (int i = 0; i < 8; ++i) {
      int k = kk0 + 16 + kbA + i;
      af.e[8 + i] = (bf16)((k < C_) ? aptr[k] : 0.0f);
    }
#pragma unroll
    for (int i = 0; i < 8; ++i) {
      int k = kk0 + kbA + i;
      bf.e[i] = (bf16)((k < C_) ? vb[((size_t)b * C_ + k) * D_ + h * HD_ + n] : 0.0f);
    }
#pragma unroll
    for (int i = 0; i < 8; ++i) {
      int k = kk0 + 16 + kbA + i;
      bf.e[8 + i] = (bf16)((k < C_) ? vb[((size_t)b * C_ + k) * D_ + h * HD_ + n] : 0.0f);
    }
    acc.v = __builtin_amdgcn_wmma_f32_16x16x32_bf16(
        false, af.v, false, bf.v, (short)0, acc.v, false, false);
  }
  const int half = lane >> 4;
  const int col  = tn * 16 + (lane & 15);
#pragma unroll
  for (int r = 0; r < 8; ++r) {
    int row = tmI * 16 + half * 8 + r;
    obuf[((size_t)b * C_ + row) * D_ + h * HD_ + col] = acc.f[r];
  }
}

// --------------------------- row-wise kernels ------------------------------
__global__ __launch_bounds__(256) void k_softmax(float* __restrict__ s, int rows, int len)
{
  int row = blockIdx.x * 8 + (threadIdx.x >> 5);
  int lane = threadIdx.x & 31;
  if (row >= rows) return;
  float* p = s + (size_t)row * len;
  float vmax = -3.0e38f;
  for (int c = lane; c < len; c += 32) vmax = fmaxf(vmax, p[c]);
  for (int o = 16; o; o >>= 1) vmax = fmaxf(vmax, __shfl_xor(vmax, o, 32));
  float sum = 0.0f;
  for (int c = lane; c < len; c += 32) { float e = __expf(p[c] - vmax); p[c] = e; sum += e; }
  for (int o = 16; o; o >>= 1) sum += __shfl_xor(sum, o, 32);
  float inv = 1.0f / sum;
  for (int c = lane; c < len; c += 32) p[c] *= inv;
}

__global__ __launch_bounds__(256) void k_l2norm(float* __restrict__ buf, int rows)
{
  int row = blockIdx.x * 8 + (threadIdx.x >> 5);
  int lane = threadIdx.x & 31;
  if (row >= rows) return;
  float* p = buf + (size_t)row * D_;
  float ss = 0.0f;
#pragma unroll
  for (int j = 0; j < 4; ++j) { float v = p[lane + j * 32]; ss += v * v; }
  for (int o = 16; o; o >>= 1) ss += __shfl_xor(ss, o, 32);
  float sc = 1.0f / fmaxf(sqrtf(ss), 1e-12f);
#pragma unroll
  for (int j = 0; j < 4; ++j) p[lane + j * 32] *= sc;
}

// out = LN(a + r) * g + be ; row length 128
__global__ __launch_bounds__(256) void k_layernorm(
    const float* __restrict__ a, const float* __restrict__ r,
    const float* __restrict__ g, const float* __restrict__ be,
    float* __restrict__ out, int rows)
{
  int row = blockIdx.x * 8 + (threadIdx.x >> 5);
  int lane = threadIdx.x & 31;
  if (row >= rows) return;
  const float* pa = a + (size_t)row * D_;
  const float* pr = r + (size_t)row * D_;
  float v[4]; float s = 0.0f, s2 = 0.0f;
#pragma unroll
  for (int j = 0; j < 4; ++j) {
    int c = lane + j * 32;
    v[j] = pa[c] + pr[c];
    s += v[j]; s2 += v[j] * v[j];
  }
  for (int o = 16; o; o >>= 1) { s += __shfl_xor(s, o, 32); s2 += __shfl_xor(s2, o, 32); }
  float mean = s * (1.0f / 128.0f);
  float var  = s2 * (1.0f / 128.0f) - mean * mean;
  float rstd = rsqrtf(var + 1e-5f);
#pragma unroll
  for (int j = 0; j < 4; ++j) {
    int c = lane + j * 32;
    out[(size_t)row * D_ + c] = (v[j] - mean) * rstd * g[c] + be[c];
  }
}

__global__ void k_colsum(const float* __restrict__ X, float* __restrict__ out,
                         int rows, int cols)
{
  int n = blockIdx.x * blockDim.x + threadIdx.x;
  if (n >= cols) return;
  float s = 0.0f;
  for (int r = 0; r < rows; ++r) s += X[(size_t)r * cols + n];
  out[n] = s;
}

// --------------------------- elementwise kernels ---------------------------
__global__ void k_zero(float* p, int n)
{ int i = blockIdx.x * blockDim.x + threadIdx.x; if (i < n) p[i] = 0.0f; }

__global__ void k_concat(const float* __restrict__ pm, const float* __restrict__ retr,
                         const float* __restrict__ x, float* __restrict__ c, int total)
{
  int i = blockIdx.x * blockDim.x + threadIdx.x;
  if (i >= total) return;
  int d = i % D_;
  int cc = (i / D_) % C_;
  int b = i / (C_ * D_);
  float v;
  if (cc < P_)            v = pm[(size_t)cc * D_ + d];
  else if (cc < P_ + S_)  v = retr[((size_t)b * S_ + (cc - P_)) * D_ + d];
  else                    v = x[((size_t)b * S_ + (cc - P_ - S_)) * D_ + d];
  c[i] = v;
}

__global__ void k_dpred(const float* __restrict__ pred, const float* __restrict__ vv,
                        float* __restrict__ dp, int n, float sc)
{ int i = blockIdx.x * blockDim.x + threadIdx.x; if (i < n) dp[i] = (pred[i] - vv[i]) * sc; }

__global__ void k_dz(const float* __restrict__ dh, const float* __restrict__ z,
                     float* __restrict__ dz, int n)
{
  int i = blockIdx.x * blockDim.x + threadIdx.x;
  if (i >= n) return;
  float zz = z[i];
  float sg = sigm(zz);
  dz[i] = dh[i] * sg * (1.0f + zz * (1.0f - sg));
}

__global__ void k_update(const float* __restrict__ w, const float* __restrict__ s,
                         const float* __restrict__ g, float* __restrict__ o, int n)
{
  int i = blockIdx.x * blockDim.x + threadIdx.x;
  if (i < n) o[i] = 0.999f * w[i] + 0.8f * s[i] - 0.3f * g[i];
}

__global__ void k_addbias(float* __restrict__ t, const float* __restrict__ b, int n, int mod)
{ int i = blockIdx.x * blockDim.x + threadIdx.x; if (i < n) t[i] += b[i % mod]; }

__global__ void k_final(const float* __restrict__ t, const float* __restrict__ r2,
                        float* __restrict__ out, int n)
{ int i = blockIdx.x * blockDim.x + threadIdx.x; if (i < n) out[i] = t[i] * sigm(r2[i]); }

// ------------------------------- ws layout ---------------------------------
// float offsets into d_ws
enum : size_t {
  OFF_Q1    = 0,                        // 131072 (q1 / later q2)
  OFF_HID   = OFF_Q1    + 131072,       // 262144 (retrieval hidden / later z1)
  OFF_RETR  = OFF_HID   + 262144,       // 131072
  OFF_CBUF  = OFF_RETR  + 131072,       // 294912
  OFF_QB    = OFF_CBUF  + 294912,       // 294912 (qb / later mha_out)
  OFF_KB    = OFF_QB    + 294912,       // 294912 (kb / later y1)
  OFF_VB    = OFF_KB    + 294912,       // 294912 (vb / later fout)
  OFF_SBUF  = OFF_VB    + 294912,       // 663552
  OFF_OBUF  = OFF_SBUF  + 663552,       // 294912 (AV out / later y2)
  OFF_FFZ   = OFF_OBUF  + 294912,       // 1179648
  OFF_T     = OFF_FFZ   + 1179648,      // 131072
  OFF_KK    = OFF_T     + 131072,       // 131072
  OFF_VV    = OFF_KK    + 131072,       // 131072
  OFF_PRED  = OFF_VV    + 131072,       // 131072
  OFF_DPRED = OFF_PRED  + 131072,       // 131072
  OFF_DH    = OFF_DPRED + 131072,       // 262144
  OFF_DZ    = OFF_DH    + 262144,       // 262144
  OFF_GL1   = OFF_DZ    + 262144,       // 32768
  OFF_GB1   = OFF_GL1   + 32768,        // 256
  OFF_GL2   = OFF_GB1   + 256,          // 32768
  OFF_GB2   = OFF_GL2   + 32768,        // 128
  OFF_NL1   = OFF_GB2   + 128,          // 32768
  OFF_NB1   = OFF_NL1   + 32768,        // 256
  OFF_NL2   = OFF_NB1   + 256,          // 32768
  OFF_NB2   = OFF_NL2   + 32768,        // 128
  OFF_Z2    = OFF_NB2   + 128,          // 262144
  OFF_R2    = OFF_Z2    + 262144        // 131072
};

// ------------------------------- launcher ----------------------------------
extern "C" void kernel_launch(void* const* d_in, const int* in_sizes, int n_in,
                              void* d_out, int out_size, void* d_ws, size_t ws_size,
                              hipStream_t stream)
{
  (void)in_sizes; (void)n_in; (void)out_size; (void)ws_size;
  const float* x        = (const float*)d_in[0];
  const float* pmem     = (const float*)d_in[1];
  const float* wq       = (const float*)d_in[2];
  const float* bq       = (const float*)d_in[3];
  const float* w_final  = (const float*)d_in[4];
  const float* b_final  = (const float*)d_in[5];
  const float* a_wq     = (const float*)d_in[6];
  const float* a_bq     = (const float*)d_in[7];
  const float* a_wk     = (const float*)d_in[8];
  const float* a_bk     = (const float*)d_in[9];
  const float* a_wv     = (const float*)d_in[10];
  const float* a_bv     = (const float*)d_in[11];
  const float* a_wo     = (const float*)d_in[12];
  const float* a_bo     = (const float*)d_in[13];
  const float* ln1_g    = (const float*)d_in[14];
  const float* ln1_b    = (const float*)d_in[15];
  const float* ln2_g    = (const float*)d_in[16];
  const float* ln2_b    = (const float*)d_in[17];
  const float* ff_w1    = (const float*)d_in[18];
  const float* ff_b1    = (const float*)d_in[19];
  const float* ff_w2    = (const float*)d_in[20];
  const float* ff_b2    = (const float*)d_in[21];
  const float* m_wk     = (const float*)d_in[22];
  const float* m_wv     = (const float*)d_in[23];
  const float* m_l1     = (const float*)d_in[24];
  const float* m_b1     = (const float*)d_in[25];
  const float* m_l2     = (const float*)d_in[26];
  const float* m_b2     = (const float*)d_in[27];
  const float* s_l1     = (const float*)d_in[28];
  const float* s_b1     = (const float*)d_in[29];
  const float* s_l2     = (const float*)d_in[30];
  const float* s_b2     = (const float*)d_in[31];
  float* out = (float*)d_out;

  float* ws    = (float*)d_ws;
  float* q1    = ws + OFF_Q1;     // also q2
  float* hid   = ws + OFF_HID;    // also z1
  float* retr  = ws + OFF_RETR;
  float* cbuf  = ws + OFF_CBUF;
  float* qb    = ws + OFF_QB;     // also mha_out
  float* kb    = ws + OFF_KB;     // also y1
  float* vb    = ws + OFF_VB;     // also fout
  float* sbuf  = ws + OFF_SBUF;
  float* obuf  = ws + OFF_OBUF;   // also y2
  float* ffz   = ws + OFF_FFZ;
  float* tbuf  = ws + OFF_T;
  float* kkb   = ws + OFF_KK;
  float* vvb   = ws + OFF_VV;
  float* predb = ws + OFF_PRED;
  float* dpred = ws + OFF_DPRED;
  float* dhb   = ws + OFF_DH;
  float* dzb   = ws + OFF_DZ;
  float* gl1   = ws + OFF_GL1;
  float* gb1   = ws + OFF_GB1;
  float* gl2   = ws + OFF_GL2;
  float* gb2   = ws + OFF_GB2;
  float* nl1   = ws + OFF_NL1;
  float* nb1   = ws + OFF_NB1;
  float* nl2   = ws + OFF_NL2;
  float* nb2   = ws + OFF_NB2;
  float* z2    = ws + OFF_Z2;
  float* r2    = ws + OFF_R2;

  const dim3 blk(256);
  auto g2 = [](int M, int N) { return dim3((N + 63) / 64, (M + 31) / 32, 1); };

  // ---- memory read: q = l2n(silu(x@wq+bq)); retrieved = mem_mlp(q) ----
  gemm_wmma<0,0,0,EPI_BIAS_SILU><<<g2(BS_, D_), blk, 0, stream>>>(x, wq, bq, q1, BS_, D_, D_, D_);
  k_l2norm<<<BS_/8, 256, 0, stream>>>(q1, BS_);
  gemm_wmma<0,0,0,EPI_BIAS><<<g2(BS_, I_), blk, 0, stream>>>(q1, m_l1, m_b1, hid, BS_, I_, D_, D_);
  gemm_wmma<0,0,1,EPI_BIAS><<<g2(BS_, D_), blk, 0, stream>>>(hid, m_l2, m_b2, retr, BS_, D_, I_, I_);

  // ---- concat [pmem | retrieved | x] -> cbuf [B,C,D] ----
  k_concat<<<(B_*C_*D_ + 255)/256, 256, 0, stream>>>(pmem, retr, x, cbuf, B_*C_*D_);

  // ---- MHA ----
  gemm_wmma<0,0,0,EPI_BIAS><<<g2(BC_, D_), blk, 0, stream>>>(cbuf, a_wq, a_bq, qb, BC_, D_, D_, D_);
  gemm_wmma<0,0,0,EPI_BIAS><<<g2(BC_, D_), blk, 0, stream>>>(cbuf, a_wk, a_bk, kb, BC_, D_, D_, D_);
  gemm_wmma<0,0,0,EPI_BIAS><<<g2(BC_, D_), blk, 0, stream>>>(cbuf, a_wv, a_bv, vb, BC_, D_, D_, D_);
  attn_scores<<<dim3(9, 9, 32), 32, 0, stream>>>(qb, kb, sbuf);
  k_softmax<<<(B_*2*C_)/8, 256, 0, stream>>>(sbuf, B_*2*C_, C_);
  attn_av<<<dim3(4, 9, 32), 32, 0, stream>>>(sbuf, vb, obuf);
  float* mha_out = qb;  // reuse
  gemm_wmma<0,0,0,EPI_BIAS><<<g2(BC_, D_), blk, 0, stream>>>(obuf, a_wo, a_bo, mha_out, BC_, D_, D_, D_);
  float* y1 = kb;       // reuse
  k_layernorm<<<BC_/8, 256, 0, stream>>>(cbuf, mha_out, ln1_g, ln1_b, y1, BC_);

  // ---- FFN + LN2 ----
  gemm_wmma<0,0,0,EPI_BIAS><<<g2(BC_, FF_), blk, 0, stream>>>(y1, ff_w1, ff_b1, ffz, BC_, FF_, D_, D_);
  float* fout = vb;     // reuse
  gemm_wmma<0,0,1,EPI_BIAS><<<g2(BC_, D_), blk, 0, stream>>>(ffz, ff_w2, ff_b2, fout, BC_, D_, FF_, FF_);
  float* y2 = obuf;     // reuse; y2 contiguous == att matrix [B, C*D]
  k_layernorm<<<BC_/8, 256, 0, stream>>>(y1, fout, ln2_g, ln2_b, y2, BC_);

  // ---- t = silu(y2.flat) @ w_final + b_final  (split-K, atomic f32) ----
  k_zero<<<(B_*SD_ + 255)/256, 256, 0, stream>>>(tbuf, B_*SD_);
  {
    dim3 grid(SD_/64, 1, 8);  // N=8192, M=16 in one row-tile, 8-way split-K
    gemm_wmma<0,0,1,EPI_ATOMIC><<<grid, blk, 0, stream>>>(
        y2, w_final, nullptr, tbuf, B_, SD_, CD_, CD_/8);
  }
  k_addbias<<<(B_*SD_ + 255)/256, 256, 0, stream>>>(tbuf, b_final, B_*SD_, SD_);

  // ---- fast-weight gradient step ----
  gemm_wmma<0,0,0,EPI_NONE><<<g2(BS_, D_), blk, 0, stream>>>(tbuf, m_wk, nullptr, kkb, BS_, D_, D_, D_);
  gemm_wmma<0,0,0,EPI_NONE><<<g2(BS_, D_), blk, 0, stream>>>(tbuf, m_wv, nullptr, vvb, BS_, D_, D_, D_);
  float* z1 = hid;      // reuse
  gemm_wmma<0,0,0,EPI_BIAS><<<g2(BS_, I_), blk, 0, stream>>>(kkb, m_l1, m_b1, z1, BS_, I_, D_, D_);
  gemm_wmma<0,0,1,EPI_BIAS><<<g2(BS_, D_), blk, 0, stream>>>(z1, m_l2, m_b2, predb, BS_, D_, I_, I_);
  k_dpred<<<(BS_*D_ + 255)/256, 256, 0, stream>>>(predb, vvb, dpred, BS_*D_, 2.0f/(float)(BS_*D_));
  // g_l2 = silu(z1)^T @ dpred   [I, D]
  gemm_wmma<1,0,1,EPI_NONE><<<g2(I_, D_), blk, 0, stream>>>(z1, dpred, nullptr, gl2, I_, D_, BS_, BS_);
  k_colsum<<<(D_ + 255)/256, 256, 0, stream>>>(dpred, gb2, BS_, D_);
  // dh = dpred @ m_l2^T   [BS, I]
  gemm_wmma<0,1,0,EPI_NONE><<<g2(BS_, I_), blk, 0, stream>>>(dpred, m_l2, nullptr, dhb, BS_, I_, D_, D_);
  k_dz<<<(BS_*I_ + 255)/256, 256, 0, stream>>>(dhb, z1, dzb, BS_*I_);
  // g_l1 = kk^T @ dz   [D, I]
  gemm_wmma<1,0,0,EPI_NONE><<<g2(D_, I_), blk, 0, stream>>>(kkb, dzb, nullptr, gl1, D_, I_, BS_, BS_);
  k_colsum<<<(I_ + 255)/256, 256, 0, stream>>>(dzb, gb1, BS_, I_);

  // ---- new = ALPHA*w + ETA*s - THETA*g ----
  k_update<<<(D_*I_ + 255)/256, 256, 0, stream>>>(m_l1, s_l1, gl1, nl1, D_*I_);
  k_update<<<1, 256, 0, stream>>>(m_b1, s_b1, gb1, nb1, I_);
  k_update<<<(I_*D_ + 255)/256, 256, 0, stream>>>(m_l2, s_l2, gl2, nl2, I_*D_);
  k_update<<<1, 256, 0, stream>>>(m_b2, s_b2, gb2, nb2, D_);

  // ---- re-retrieval with updated memory + gating ----
  float* q2 = q1;       // reuse
  gemm_wmma<0,0,0,EPI_BIAS_SILU><<<g2(BS_, D_), blk, 0, stream>>>(tbuf, wq, bq, q2, BS_, D_, D_, D_);
  k_l2norm<<<BS_/8, 256, 0, stream>>>(q2, BS_);
  gemm_wmma<0,0,0,EPI_BIAS><<<g2(BS_, I_), blk, 0, stream>>>(q2, nl1, nb1, z2, BS_, I_, D_, D_);
  gemm_wmma<0,0,1,EPI_BIAS><<<g2(BS_, D_), blk, 0, stream>>>(z2, nl2, nb2, r2, BS_, D_, I_, I_);
  k_final<<<(BS_*D_ + 255)/256, 256, 0, stream>>>(tbuf, r2, out, BS_*D_);
}
